// RNN_69483980915386
// MI455X (gfx1250) — matrix operand order repaired and checked by reading, not compiled, not run on hardware
//
#include <hip/hip_runtime.h>

// ---------------------------------------------------------------------------
// 4-layer tanh RNN scan, persistent-workgroup design for MI455X (gfx1250).
//   B=64, S=2048, D=512, L=4.  A = [x_t ; h_l] (16x1024), W = [Wh;Uh] (1024x512)
//   - 4 workgroups, one per 16-row batch slab (batch rows are independent)
//   - 32 waves/WG, each wave owns one 16-column output tile
//   - weights repacked once into WMMA B-fragment order (bf16); streamed from L2
//     via a ROLLED, ping-pong double-buffered K loop (no rotation movs,
//     partial loadcnt/dscnt waits)
//   - per-layer A matrix is one contiguous 16x1024 LDS tile: [input | h[l]]
//   - fp32 accum via v_wmma_f32_16x16x32_bf16; activation via hw v_tanh_f32
// ---------------------------------------------------------------------------

typedef __attribute__((ext_vector_type(16))) __bf16 v16bf;
typedef __attribute__((ext_vector_type(8)))  __bf16 v8bf;
typedef __attribute__((ext_vector_type(8)))  float  v8f;
typedef __attribute__((ext_vector_type(4)))  float  v4f;

#define B_   64
#define S_   2048
#define D_   512
#define L_   4
#define AW_  (2 * D_)      // 1024: width of a layer's A tile in LDS
#define NT_  (D_ / 16)     // 32 column tiles
#define KT_  (2 * D_ / 32) // 32 K tiles of 32 per layer

__device__ __forceinline__ float fast_tanh(float v) {
#if __has_builtin(__builtin_amdgcn_tanhf)
  return __builtin_amdgcn_tanhf(v);       // v_tanh_f32 (gfx1250 TRANS op)
#elif __has_builtin(__builtin_amdgcn_tanh_f32)
  return __builtin_amdgcn_tanh_f32(v);
#else
  return tanhf(v);
#endif
}

// -------- prep: repack [Wh;Uh] fp32 -> bf16 in B-fragment order --------------
// pk layout: [L][NT][KT][lane(32)][16 halves]
//   lane <16 : col = nt*16 + lane,      K = kt*32 + (h&7) + (h>>3)*16      (h = 0..15)
//   lane>=16 : col = nt*16 + lane - 16, K = kt*32 + 8 + (h&7) + (h>>3)*16
// K 0..511 -> Wh rows, K 512..1023 -> Uh rows (matches A = [input | h]).
__global__ void rnn_pack_weights(const float* __restrict__ Wh,
                                 const float* __restrict__ Uh,
                                 __bf16* __restrict__ pk) {
  size_t idx = blockIdx.x * (size_t)blockDim.x + threadIdx.x;
  const size_t total = (size_t)L_ * NT_ * KT_ * 32 * 16;
  if (idx >= total) return;
  int h    = (int)(idx & 15);
  int lane = (int)((idx >> 4) & 31);
  int kt   = (int)((idx >> 9) & 31);
  int nt   = (int)((idx >> 14) & 31);
  int l    = (int)(idx >> 19);
  int col  = nt * 16 + (lane & 15);
  int k    = kt * 32 + ((lane >> 4) ? 8 : 0) + (h & 7) + ((h >> 3) * 16);
  float w  = (k < D_) ? Wh[((size_t)l * D_ + k) * D_ + col]
                      : Uh[((size_t)l * D_ + (k - D_)) * D_ + col];
  pk[idx] = (__bf16)w;
}

__device__ __forceinline__ v16bf cat8(v8bf lo, v8bf hi) {
  // lands in adjacent VGPRs after the two ds_load_b128 -> zero-cost concat
  return __builtin_shufflevector(lo, hi, 0, 1, 2, 3, 4, 5, 6, 7,
                                         8, 9, 10, 11, 12, 13, 14, 15);
}

#define WMMA_BF16(av, bv, cc)                                            \
  __builtin_amdgcn_wmma_f32_16x16x32_bf16(false, (av), false, (bv),      \
                                          (short)0, (cc), false, false)

// One full 1024-wide K pass (32 K-tiles) over a contiguous 16x1024 A tile.
// ROLLED loop with ping-pong buffers: each buffer is reloaded immediately
// after the WMMA that consumes it, so no rotation movs and each WMMA waits
// only for loads issued one full iteration earlier (one newer set in flight).
//   ap: this lane's A base in LDS  (row-major bf16, row*AW_ + laneKoff)
//   wp: this lane's packed B base  (+512 halves per K-tile)
__device__ __forceinline__ v8f kpass(const __bf16* ap,
                                     const __bf16* __restrict__ wp,
                                     v8f c) {
  // Opaque ZERO OFFSET (not an opaque pointer): blocks LICM from hoisting the
  // t-invariant prologue loads out of the scan loop (which previously spilled
  // them to scratch), while keeping the pointer's provenance so the loads stay
  // global_load_b128 (LOADcnt only) instead of regressing to flat_load.
  int zoff = 0;
  asm volatile("" : "+v"(zoff));
  wp += zoff;
  v8bf  xlo = *(const v8bf*)(ap);           // ds_load_b128
  v8bf  xhi = *(const v8bf*)(ap + 16);      // ds_load_b128
  v16bf xb  = *(const v16bf*)(wp);          // 2x global_load_b128 (L2 hit)
  v8bf  ylo = *(const v8bf*)(ap + 32);
  v8bf  yhi = *(const v8bf*)(ap + 48);
  v16bf yb  = *(const v16bf*)(wp + 512);
#pragma unroll 1
  for (int kt = 0; kt < KT_ - 2; kt += 2) {
    c   = WMMA_BF16(cat8(xlo, xhi), xb, c);
    xlo = *(const v8bf*)(ap + 64);          // refill dead x-buffer (kt+2)
    xhi = *(const v8bf*)(ap + 80);
    xb  = *(const v16bf*)(wp + 1024);
    c   = WMMA_BF16(cat8(ylo, yhi), yb, c);
    ylo = *(const v8bf*)(ap + 96);          // refill dead y-buffer (kt+3)
    yhi = *(const v8bf*)(ap + 112);
    yb  = *(const v16bf*)(wp + 1536);
    ap += 64; wp += 1024;
  }
  c = WMMA_BF16(cat8(xlo, xhi), xb, c);
  return WMMA_BF16(cat8(ylo, yhi), yb, c);
}

__launch_bounds__(1024, 1)
__global__ void rnn_scan(const float* __restrict__ x,      // [B,S,D]
                         const __bf16* __restrict__ pk,    // packed weights
                         const float* __restrict__ bias,   // [L,D]
                         float* __restrict__ out) {        // [B,S,D]
  // Per-layer A tile, contiguous: cols 0..511 = layer input, 512..1023 = h[l]
  __shared__ __align__(16) __bf16 abuf[L_][16 * AW_];   // 4 x 32 KB = 128 KB

  const int tid  = (int)threadIdx.x;
  const int wave = tid >> 5;          // 0..31  (wave32)
  const int lane = tid & 31;
  const int b0   = (int)blockIdx.x * 16;            // batch slab base
  const int ncol = wave * 16 + (lane & 15);         // this lane's output col
  const int mhi  = (lane >> 4) ? 8 : 0;             // C/D row offset
  // A-fragment per-lane base offset into a row-major 16 x AW_ LDS tile:
  // lane<16 -> row=lane, K chunks {0..7, 16..23}; lane>=16 -> row=lane-16,
  // K chunks {8..15, 24..31}  (16-bit A 16x32 layout)
  const int abase = (lane & 15) * AW_ + ((lane >> 4) ? 8 : 0);

  // zero all A tiles (h0 = 0; input halves get overwritten before use)
  for (int i = tid; i < L_ * 16 * AW_; i += (int)blockDim.x)
    (&abuf[0][0])[i] = (__bf16)0.0f;

  // per-layer bias for my column + per-layer packed-weight base (scan-invariant)
  float bl[L_];
  const __bf16* wbase[L_];
#pragma unroll
  for (int l = 0; l < L_; ++l) {
    bl[l] = bias[l * D_ + ncol];
    wbase[l] = pk + (((size_t)l * NT_ + wave) * KT_) * (32 * 16) + lane * 16;
  }

  __syncthreads();

  for (int t = 0; t < S_; ++t) {
    // ---- stage x[:, t, :] into abuf[0] input half (fp32 -> bf16) ----
    {
      int row = tid >> 6;                 // 16 rows, 64 threads per row
      int cb  = (tid & 63) * 8;           // 8 consecutive columns each
      const float* src = x + ((size_t)(b0 + row) * S_ + t) * D_ + cb;
      v4f a  = *(const v4f*)(src);
      v4f b4 = *(const v4f*)(src + 4);
      // warm L2/WGP$ for the next time step while the 4 layers compute
      // (speculative prefetch: an OOB address at the tail is silently dropped)
      __builtin_prefetch(src + D_, 0, 3);
      __bf16* dst = &abuf[0][0] + row * AW_ + cb;
      dst[0] = (__bf16)a.x;  dst[1] = (__bf16)a.y;
      dst[2] = (__bf16)a.z;  dst[3] = (__bf16)a.w;
      dst[4] = (__bf16)b4.x; dst[5] = (__bf16)b4.y;
      dst[6] = (__bf16)b4.z; dst[7] = (__bf16)b4.w;
    }
    __syncthreads();

#pragma unroll
    for (int l = 0; l < L_; ++l) {
      v8f c;
#pragma unroll
      for (int r = 0; r < 8; ++r) c[r] = bl[l];

      // full K=1024 pass over contiguous [input | h[l]] tile
      c = kpass(&abuf[l][0] + abase, wbase[l], c);

      __syncthreads();   // everyone done reading abuf[l]

      // tanh; write new h[l] and next layer's input; last layer -> out
#pragma unroll
      for (int r = 0; r < 8; ++r) {
        float v = fast_tanh(c[r]);
        int m = r + mhi;
        abuf[l][m * AW_ + D_ + ncol] = (__bf16)v;        // new h[l]
        if (l < L_ - 1)
          abuf[l + 1][m * AW_ + ncol] = (__bf16)v;       // input to layer l+1
        else
          out[((size_t)(b0 + m) * S_ + t) * D_ + ncol] = v;
      }
      __syncthreads();   // writes visible before layer l+1 (or next t) reads
    }
  }
}

extern "C" void kernel_launch(void* const* d_in, const int* in_sizes, int n_in,
                              void* d_out, int out_size, void* d_ws, size_t ws_size,
                              hipStream_t stream) {
  const float* x  = (const float*)d_in[0];   // [B,S,D] fp32
  const float* Wh = (const float*)d_in[1];   // [L,D,D] fp32
  const float* Uh = (const float*)d_in[2];   // [L,D,D] fp32
  const float* b  = (const float*)d_in[3];   // [L,D]   fp32

  __bf16* pk = (__bf16*)d_ws;                // 4 MB packed bf16 weights

  const size_t total = (size_t)L_ * NT_ * KT_ * 32 * 16;
  rnn_pack_weights<<<(unsigned)((total + 255) / 256), 256, 0, stream>>>(Wh, Uh, pk);

  rnn_scan<<<B_ / 16, 1024, 0, stream>>>(x, pk, b, (float*)d_out);
}
